// CrystalGraphNetwork_28192165331135
// MI455X (gfx1250) — matrix operand order repaired and checked by reading, not compiled, not run on hardware
//
#include <hip/hip_runtime.h>

typedef __attribute__((ext_vector_type(16))) _Float16 v16h;
typedef __attribute__((ext_vector_type(8)))  _Float16 v8h;
typedef __attribute__((ext_vector_type(8)))  float    v8f;

#define HID 128

__device__ __forceinline__ float silu_f(float x) {
    return x / (1.0f + __expf(-x));
}

__device__ __forceinline__ v16h wmma_a_frag(const _Float16* base, int row,
                                            int stride, int kt, int koffA) {
    v8h lo = *(const v8h*)&base[row * stride + kt * 32 + koffA];
    v8h hi = *(const v8h*)&base[row * stride + kt * 32 + koffA + 8];
    v16h a;
    #pragma unroll
    for (int i = 0; i < 8; ++i) { a[i] = lo[i]; a[i + 8] = hi[i]; }
    return a;
}

// ---------------------------------------------------------------------------
// Zero a float buffer
// ---------------------------------------------------------------------------
__global__ void zero_kernel(float* __restrict__ p, int n) {
    int i = blockIdx.x * blockDim.x + threadIdx.x;
    if (i < n) p[i] = 0.0f;
}

// ---------------------------------------------------------------------------
// Edge features: bessel basis (8) + real SH l<=2 (9), padded to 32 f16
// ---------------------------------------------------------------------------
__global__ void edge_feat_kernel(const float* __restrict__ edges,
                                 const float* __restrict__ freqs,
                                 _Float16* __restrict__ ef, int E) {
    int e = blockIdx.x * blockDim.x + threadIdx.x;
    if (e >= E) return;
    float ex = edges[3 * e + 0], ey = edges[3 * e + 1], ez = edges[3 * e + 2];
    float d = sqrtf(ex * ex + ey * ey + ez * ez);
    float scaled = fminf(fmaxf(d * 0.25f, 0.0f), 1.0f);   // d / CUTOFF(=4)
    _Float16* o = ef + (size_t)e * 32;
    #pragma unroll
    for (int nb = 0; nb < 8; ++nb) {
        float x = scaled * freqs[nb];
        o[nb] = (_Float16)(__sinf(x) / x);
    }
    float inv = 1.0f / (d + 1e-7f);
    float x = ex * inv, y = ey * inv, z = ez * inv;
    const float c0 = 0.28209479177387814f, c1 = 0.4886025119029199f;
    const float c2a = 1.0925484305920792f, c2b = 0.31539156525252005f;
    const float c2c = 0.5462742152960396f;
    o[8]  = (_Float16)c0;
    o[9]  = (_Float16)(c1 * y);
    o[10] = (_Float16)(c1 * z);
    o[11] = (_Float16)(c1 * x);
    o[12] = (_Float16)(c2a * x * y);
    o[13] = (_Float16)(c2a * y * z);
    o[14] = (_Float16)(c2b * (3.0f * z * z - 1.0f));
    o[15] = (_Float16)(c2a * x * z);
    o[16] = (_Float16)(c2c * (x * x - y * y));
    #pragma unroll
    for (int j = 17; j < 32; ++j) o[j] = (_Float16)0.0f;
}

// ---------------------------------------------------------------------------
// Pack fp32 weight W[K x 128] into per-lane WMMA B-fragment layout (f16).
// Fragment (nt, kt): lane L holds W[kt*32 + (L<16?0:16) + e][nt*16 + (L&15)]
// for e = 0..15 contiguously -> one 32B load per lane at use time.
// ---------------------------------------------------------------------------
__global__ void pack_weight_kernel(const float* __restrict__ W,
                                   _Float16* __restrict__ out,
                                   int Korig, int Kpad) {
    int gid = blockIdx.x * blockDim.x + threadIdx.x;
    int nKT = Kpad >> 5;
    int total = 8 * nKT * 32;                 // 8 n-tiles of 16 cols (N=128)
    if (gid >= total) return;
    int lane = gid & 31;
    int t = gid >> 5;
    int kt = t % nKT;
    int nt = t / nKT;
    int nn = nt * 16 + (lane & 15);
    int kbase = kt * 32 + ((lane < 16) ? 0 : 16);
    _Float16* o = out + ((size_t)(nt * nKT + kt) * 32 + lane) * 16;
    #pragma unroll
    for (int e2 = 0; e2 < 16; ++e2) {
        int k = kbase + e2;
        float v = (k < Korig) ? W[(size_t)k * HID + nn] : 0.0f;
        o[e2] = (_Float16)v;
    }
}

// ---------------------------------------------------------------------------
// Fused message MLP + segment-sum scatter.
// Block = 64 edges, 8 waves. Each wave owns a 2x2 register tile:
//   M-pair = wave>>2 (rows 32*mp .. 32*mp+31), N-pair = wave&3 (cols 32*np..).
// Per k-step: 2 A-frag LDS loads + 2 B-frag global loads -> 4 WMMAs.
// GEMM1: [64 x 288] x [288 x 128] (9 k-steps), SiLU, f16 -> LDS
// GEMM2: [64 x 128] x [128 x 128] (4 k-steps), +bias, atomicAdd into agg
// ---------------------------------------------------------------------------
__global__ void __launch_bounds__(256)
msg_kernel(const float* __restrict__ h,
           const int* __restrict__ senders,
           const int* __restrict__ receivers,
           const _Float16* __restrict__ ef,
           const _Float16* __restrict__ pw1, const float* __restrict__ mb1,
           const _Float16* __restrict__ pw2, const float* __restrict__ mb2,
           float* __restrict__ agg, int E) {
    __shared__ __align__(16) _Float16 smA[64 * 288];
    __shared__ __align__(16) _Float16 smM[64 * HID];
    __shared__ int smRecv[64];

    int tid = threadIdx.x;
    // ---- stage A = [h_s | h_r | ef] as f16 (64 rows x 4 threads/row) ----
    {
        int row = tid >> 2, c = tid & 3;
        int e = blockIdx.x * 64 + row;
        bool valid = (e < E);
        int ec = valid ? e : (E - 1);
        int s = senders[ec], r = receivers[ec];
        if (c == 0) smRecv[row] = valid ? r : -1;
        const float4* hs4 = (const float4*)(h + (size_t)s * HID);
        const float4* hr4 = (const float4*)(h + (size_t)r * HID);
        #pragma unroll
        for (int q = 0; q < 4; ++q) {             // 4 x v8h chunks = 32 halfs
            v8h va, vb;
            #pragma unroll
            for (int p = 0; p < 2; ++p) {          // 2 float4 per v8h
                float4 fa = valid ? hs4[c * 8 + q * 2 + p]
                                  : make_float4(0.f, 0.f, 0.f, 0.f);
                float4 fb = valid ? hr4[c * 8 + q * 2 + p]
                                  : make_float4(0.f, 0.f, 0.f, 0.f);
                va[p * 4 + 0] = (_Float16)fa.x; va[p * 4 + 1] = (_Float16)fa.y;
                va[p * 4 + 2] = (_Float16)fa.z; va[p * 4 + 3] = (_Float16)fa.w;
                vb[p * 4 + 0] = (_Float16)fb.x; vb[p * 4 + 1] = (_Float16)fb.y;
                vb[p * 4 + 2] = (_Float16)fb.z; vb[p * 4 + 3] = (_Float16)fb.w;
            }
            *(v8h*)&smA[row * 288 + c * 32 + q * 8]       = va;
            *(v8h*)&smA[row * 288 + 128 + c * 32 + q * 8] = vb;
        }
        v8h ve = {};
        if (valid) ve = *(const v8h*)&ef[(size_t)ec * 32 + c * 8];
        *(v8h*)&smA[row * 288 + 256 + c * 8] = ve;
    }
    __syncthreads();

    int wave = tid >> 5, lane = tid & 31;
    int m = lane & 15;
    int koffA = (lane < 16) ? 0 : 16;
    int mp = wave >> 2;                    // 0..1 : M pair
    int np = wave & 3;                     // 0..3 : N pair
    int rbase = mp * 32 + m;               // A row for M tile 0
    int mrowBase = (lane >> 4) << 3;       // C/D rows: rr + 8*(lane/16)

    // ---- GEMM1: K = 288, 2x2 register blocking ----
    v8f acc[2][2] = {};
    #pragma unroll
    for (int kt = 0; kt < 9; ++kt) {
        v16h a0 = wmma_a_frag(smA, rbase,      288, kt, koffA);
        v16h a1 = wmma_a_frag(smA, rbase + 16, 288, kt, koffA);
        v16h b0 = *(const v16h*)&pw1[(((size_t)(np * 2 + 0) * 9 + kt) * 32 + lane) * 16];
        v16h b1 = *(const v16h*)&pw1[(((size_t)(np * 2 + 1) * 9 + kt) * 32 + lane) * 16];
        acc[0][0] = __builtin_amdgcn_wmma_f32_16x16x32_f16(false, a0, false, b0, (short)0, acc[0][0], false, false);
        acc[0][1] = __builtin_amdgcn_wmma_f32_16x16x32_f16(false, a0, false, b1, (short)0, acc[0][1], false, false);
        acc[1][0] = __builtin_amdgcn_wmma_f32_16x16x32_f16(false, a1, false, b0, (short)0, acc[1][0], false, false);
        acc[1][1] = __builtin_amdgcn_wmma_f32_16x16x32_f16(false, a1, false, b1, (short)0, acc[1][1], false, false);
    }
    #pragma unroll
    for (int tm = 0; tm < 2; ++tm) {
        #pragma unroll
        for (int tn = 0; tn < 2; ++tn) {
            int n = np * 32 + tn * 16 + m;
            float bv = mb1[n];
            #pragma unroll
            for (int rr = 0; rr < 8; ++rr) {
                int mr = mp * 32 + tm * 16 + mrowBase + rr;
                smM[mr * HID + n] = (_Float16)silu_f(acc[tm][tn][rr] + bv);
            }
        }
    }
    __syncthreads();

    // ---- GEMM2: K = 128, 2x2 register blocking ----
    v8f acc2[2][2] = {};
    #pragma unroll
    for (int kt = 0; kt < 4; ++kt) {
        v16h a0 = wmma_a_frag(smM, rbase,      HID, kt, koffA);
        v16h a1 = wmma_a_frag(smM, rbase + 16, HID, kt, koffA);
        v16h b0 = *(const v16h*)&pw2[(((size_t)(np * 2 + 0) * 4 + kt) * 32 + lane) * 16];
        v16h b1 = *(const v16h*)&pw2[(((size_t)(np * 2 + 1) * 4 + kt) * 32 + lane) * 16];
        acc2[0][0] = __builtin_amdgcn_wmma_f32_16x16x32_f16(false, a0, false, b0, (short)0, acc2[0][0], false, false);
        acc2[0][1] = __builtin_amdgcn_wmma_f32_16x16x32_f16(false, a0, false, b1, (short)0, acc2[0][1], false, false);
        acc2[1][0] = __builtin_amdgcn_wmma_f32_16x16x32_f16(false, a1, false, b0, (short)0, acc2[1][0], false, false);
        acc2[1][1] = __builtin_amdgcn_wmma_f32_16x16x32_f16(false, a1, false, b1, (short)0, acc2[1][1], false, false);
    }
    #pragma unroll
    for (int tm = 0; tm < 2; ++tm) {
        #pragma unroll
        for (int tn = 0; tn < 2; ++tn) {
            int n = np * 32 + tn * 16 + m;
            float bv = mb2[n];
            #pragma unroll
            for (int rr = 0; rr < 8; ++rr) {
                int mr = mp * 32 + tm * 16 + mrowBase + rr;
                int node = smRecv[mr];
                if (node >= 0)
                    atomicAdd(&agg[(size_t)node * HID + n], acc2[tm][tn][rr] + bv);
            }
        }
    }
}

// ---------------------------------------------------------------------------
// Node update MLP: h' = silu([h|agg] @ uw1 + ub1) @ uw2 + ub2
// Block = 64 nodes, 8 waves, 2x2 register tiles. K1 = 256, K2 = 128.
// ---------------------------------------------------------------------------
__global__ void __launch_bounds__(256)
upd_kernel(const float* __restrict__ h, const float* __restrict__ agg,
           const _Float16* __restrict__ pw1, const float* __restrict__ ub1,
           const _Float16* __restrict__ pw2, const float* __restrict__ ub2,
           float* __restrict__ hout, int Nn) {
    __shared__ __align__(16) _Float16 smA[64 * 256];
    __shared__ __align__(16) _Float16 smM[64 * HID];

    int tid = threadIdx.x;
    {
        int row = tid >> 2, c = tid & 3;
        int node = blockIdx.x * 64 + row;
        bool valid = (node < Nn);
        int nc = valid ? node : (Nn - 1);
        const float4* ha4 = (const float4*)(h   + (size_t)nc * HID);
        const float4* hb4 = (const float4*)(agg + (size_t)nc * HID);
        #pragma unroll
        for (int q = 0; q < 4; ++q) {
            v8h va, vb;
            #pragma unroll
            for (int p = 0; p < 2; ++p) {
                float4 fa = valid ? ha4[c * 8 + q * 2 + p]
                                  : make_float4(0.f, 0.f, 0.f, 0.f);
                float4 fb = valid ? hb4[c * 8 + q * 2 + p]
                                  : make_float4(0.f, 0.f, 0.f, 0.f);
                va[p * 4 + 0] = (_Float16)fa.x; va[p * 4 + 1] = (_Float16)fa.y;
                va[p * 4 + 2] = (_Float16)fa.z; va[p * 4 + 3] = (_Float16)fa.w;
                vb[p * 4 + 0] = (_Float16)fb.x; vb[p * 4 + 1] = (_Float16)fb.y;
                vb[p * 4 + 2] = (_Float16)fb.z; vb[p * 4 + 3] = (_Float16)fb.w;
            }
            *(v8h*)&smA[row * 256 + c * 32 + q * 8]       = va;
            *(v8h*)&smA[row * 256 + 128 + c * 32 + q * 8] = vb;
        }
    }
    __syncthreads();

    int wave = tid >> 5, lane = tid & 31;
    int m = lane & 15;
    int koffA = (lane < 16) ? 0 : 16;
    int mp = wave >> 2;
    int np = wave & 3;
    int rbase = mp * 32 + m;
    int mrowBase = (lane >> 4) << 3;

    v8f acc[2][2] = {};
    #pragma unroll
    for (int kt = 0; kt < 8; ++kt) {
        v16h a0 = wmma_a_frag(smA, rbase,      256, kt, koffA);
        v16h a1 = wmma_a_frag(smA, rbase + 16, 256, kt, koffA);
        v16h b0 = *(const v16h*)&pw1[(((size_t)(np * 2 + 0) * 8 + kt) * 32 + lane) * 16];
        v16h b1 = *(const v16h*)&pw1[(((size_t)(np * 2 + 1) * 8 + kt) * 32 + lane) * 16];
        acc[0][0] = __builtin_amdgcn_wmma_f32_16x16x32_f16(false, a0, false, b0, (short)0, acc[0][0], false, false);
        acc[0][1] = __builtin_amdgcn_wmma_f32_16x16x32_f16(false, a0, false, b1, (short)0, acc[0][1], false, false);
        acc[1][0] = __builtin_amdgcn_wmma_f32_16x16x32_f16(false, a1, false, b0, (short)0, acc[1][0], false, false);
        acc[1][1] = __builtin_amdgcn_wmma_f32_16x16x32_f16(false, a1, false, b1, (short)0, acc[1][1], false, false);
    }
    #pragma unroll
    for (int tm = 0; tm < 2; ++tm) {
        #pragma unroll
        for (int tn = 0; tn < 2; ++tn) {
            int n = np * 32 + tn * 16 + m;
            float bv = ub1[n];
            #pragma unroll
            for (int rr = 0; rr < 8; ++rr) {
                int mr = mp * 32 + tm * 16 + mrowBase + rr;
                smM[mr * HID + n] = (_Float16)silu_f(acc[tm][tn][rr] + bv);
            }
        }
    }
    __syncthreads();

    v8f acc2[2][2] = {};
    #pragma unroll
    for (int kt = 0; kt < 4; ++kt) {
        v16h a0 = wmma_a_frag(smM, rbase,      HID, kt, koffA);
        v16h a1 = wmma_a_frag(smM, rbase + 16, HID, kt, koffA);
        v16h b0 = *(const v16h*)&pw2[(((size_t)(np * 2 + 0) * 4 + kt) * 32 + lane) * 16];
        v16h b1 = *(const v16h*)&pw2[(((size_t)(np * 2 + 1) * 4 + kt) * 32 + lane) * 16];
        acc2[0][0] = __builtin_amdgcn_wmma_f32_16x16x32_f16(false, a0, false, b0, (short)0, acc2[0][0], false, false);
        acc2[0][1] = __builtin_amdgcn_wmma_f32_16x16x32_f16(false, a0, false, b1, (short)0, acc2[0][1], false, false);
        acc2[1][0] = __builtin_amdgcn_wmma_f32_16x16x32_f16(false, a1, false, b0, (short)0, acc2[1][0], false, false);
        acc2[1][1] = __builtin_amdgcn_wmma_f32_16x16x32_f16(false, a1, false, b1, (short)0, acc2[1][1], false, false);
    }
    #pragma unroll
    for (int tm = 0; tm < 2; ++tm) {
        #pragma unroll
        for (int tn = 0; tn < 2; ++tn) {
            int n = np * 32 + tn * 16 + m;
            float bv = ub2[n];
            #pragma unroll
            for (int rr = 0; rr < 8; ++rr) {
                int mr = mp * 32 + tm * 16 + mrowBase + rr;
                int nd = blockIdx.x * 64 + mr;
                if (nd < Nn) hout[(size_t)nd * HID + n] = acc2[tm][tn][rr] + bv;
            }
        }
    }
}

// ---------------------------------------------------------------------------
// Graph pooling: segment sums + counts
// ---------------------------------------------------------------------------
__global__ void pool_kernel(const float* __restrict__ h,
                            const int* __restrict__ gids,
                            float* __restrict__ sums,
                            float* __restrict__ counts, int Nn) {
    int idx = blockIdx.x * blockDim.x + threadIdx.x;
    int i = idx >> 7, c = idx & 127;
    if (i >= Nn) return;
    int g = gids[i];
    atomicAdd(&sums[(size_t)g * HID + c], h[(size_t)i * HID + c]);
    if (c == 0) atomicAdd(&counts[g], 1.0f);
}

// ---------------------------------------------------------------------------
// Output head: mean, silu(gf @ ow1 + ob1) @ ow2 + ob2   (per-graph block)
// ---------------------------------------------------------------------------
__global__ void __launch_bounds__(128)
out_kernel(const float* __restrict__ sums, const float* __restrict__ counts,
           const float* __restrict__ ow1, const float* __restrict__ ob1,
           const float* __restrict__ ow2, const float* __restrict__ ob2,
           float* __restrict__ out) {
    __shared__ float gf[HID];
    __shared__ float red[HID];
    int g = blockIdx.x, j = threadIdx.x;
    float cnt = fmaxf(counts[g], 1.0f);
    gf[j] = sums[(size_t)g * HID + j] / cnt;
    __syncthreads();
    float acc = ob1[j];
    #pragma unroll 4
    for (int c = 0; c < HID; ++c) acc += gf[c] * ow1[c * HID + j];
    red[j] = silu_f(acc) * ow2[j];
    __syncthreads();
    for (int s = 64; s > 0; s >>= 1) {
        if (j < s) red[j] += red[j + s];
        __syncthreads();
    }
    if (j == 0) out[g] = red[0] + ob2[0];
}

// ---------------------------------------------------------------------------
// Host launcher
// ---------------------------------------------------------------------------
extern "C" void kernel_launch(void* const* d_in, const int* in_sizes, int n_in,
                              void* d_out, int out_size, void* d_ws, size_t ws_size,
                              hipStream_t stream) {
    (void)n_in; (void)ws_size;
    const int H = HID, L = 4;
    const int N = in_sizes[0] / H;   // 50000
    const int E = in_sizes[2];       // 800000
    const int G = out_size;          // 256

    const float* nodes   = (const float*)d_in[0];
    const float* edges   = (const float*)d_in[1];
    const int*   senders = (const int*)d_in[2];
    const int*   recvs   = (const int*)d_in[3];
    const int*   gids    = (const int*)d_in[4];
    const float* freqs   = (const float*)d_in[5];
    // params (insertion order): per layer {mw1,mb1,mw2,mb2,uw1,ub1,uw2,ub2}
    const float *mw1[4], *mb1[4], *mw2[4], *mb2[4];
    const float *uw1[4], *ub1[4], *uw2[4], *ub2[4];
    for (int l = 0; l < L; ++l) {
        int b = 6 + 8 * l;
        mw1[l] = (const float*)d_in[b + 0];
        mb1[l] = (const float*)d_in[b + 1];
        mw2[l] = (const float*)d_in[b + 2];
        mb2[l] = (const float*)d_in[b + 3];
        uw1[l] = (const float*)d_in[b + 4];
        ub1[l] = (const float*)d_in[b + 5];
        uw2[l] = (const float*)d_in[b + 6];
        ub2[l] = (const float*)d_in[b + 7];
    }
    const float* ow1 = (const float*)d_in[38];
    const float* ob1 = (const float*)d_in[39];
    const float* ow2 = (const float*)d_in[40];
    const float* ob2 = (const float*)d_in[41];

    // ---- workspace carve-out (256B aligned) ----
    char* ws = (char*)d_ws;
    size_t off = 0;
    auto carve = [&](size_t bytes) -> void* {
        void* p = ws + off;
        off = (off + bytes + 255) & ~(size_t)255;
        return p;
    };
    float*    h0     = (float*)carve((size_t)N * H * 4);
    float*    h1     = (float*)carve((size_t)N * H * 4);
    float*    agg    = (float*)carve((size_t)N * H * 4);
    float*    sums   = (float*)carve((size_t)G * H * 4);
    float*    counts = (float*)carve((size_t)G * 4);
    _Float16* ef     = (_Float16*)carve((size_t)E * 32 * 2);
    _Float16 *pw1[4], *pw2[4], *pu1[4], *pu2[4];
    for (int l = 0; l < L; ++l) {
        pw1[l] = (_Float16*)carve((size_t)288 * H * 2);
        pw2[l] = (_Float16*)carve((size_t)128 * H * 2);
        pu1[l] = (_Float16*)carve((size_t)256 * H * 2);
        pu2[l] = (_Float16*)carve((size_t)128 * H * 2);
    }

    // ---- precompute: edge features + weight packing ----
    edge_feat_kernel<<<(E + 255) / 256, 256, 0, stream>>>(edges, freqs, ef, E);
    for (int l = 0; l < L; ++l) {
        pack_weight_kernel<<<(8 * 288 + 255) / 256, 256, 0, stream>>>(mw1[l], pw1[l], 273, 288);
        pack_weight_kernel<<<(8 * 128 + 255) / 256, 256, 0, stream>>>(mw2[l], pw2[l], 128, 128);
        pack_weight_kernel<<<(8 * 256 + 255) / 256, 256, 0, stream>>>(uw1[l], pu1[l], 256, 256);
        pack_weight_kernel<<<(8 * 128 + 255) / 256, 256, 0, stream>>>(uw2[l], pu2[l], 128, 128);
    }

    // ---- message-passing layers ----
    const float* hcur = nodes;
    for (int l = 0; l < L; ++l) {
        zero_kernel<<<((N * H) + 255) / 256, 256, 0, stream>>>(agg, N * H);
        msg_kernel<<<(E + 63) / 64, 256, 0, stream>>>(
            hcur, senders, recvs, ef, pw1[l], mb1[l], pw2[l], mb2[l], agg, E);
        float* hnext = (l & 1) ? h1 : h0;
        upd_kernel<<<(N + 63) / 64, 256, 0, stream>>>(
            hcur, agg, pu1[l], ub1[l], pu2[l], ub2[l], hnext, N);
        hcur = hnext;
    }

    // ---- pooling + output head ----
    zero_kernel<<<((G * H) + 255) / 256, 256, 0, stream>>>(sums, G * H);
    zero_kernel<<<(G + 255) / 256, 256, 0, stream>>>(counts, G);
    pool_kernel<<<((size_t)N * H + 255) / 256, 256, 0, stream>>>(hcur, gids, sums, counts, N);
    out_kernel<<<G, 128, 0, stream>>>(sums, counts, ow1, ob1, ow2, ob2, (float*)d_out);
}
